// PonitaFC_83657372992127
// MI455X (gfx1250) — matrix operand order repaired and checked by reading, not compiled, not run on hardware
//
#include <hip/hip_runtime.h>
#include <hip/hip_bf16.h>
#include <math.h>

typedef __attribute__((ext_vector_type(16))) __bf16 v16bf;
typedef __attribute__((ext_vector_type(8)))  float  v8f;
typedef __attribute__((ext_vector_type(4)))  unsigned int u32x4;

union FragBF {
  v16bf v;
  unsigned short s[16];
  u32x4 u[2];
};

__device__ __forceinline__ unsigned short f2bf(float x) {
  unsigned int u = __float_as_uint(x);
  unsigned int r = u + 0x7fffu + ((u >> 16) & 1u);
  return (unsigned short)(r >> 16);
}
__device__ __forceinline__ float gelu_f(float x) {
  return 0.5f * x * (1.0f + erff(x * 0.70710678118654752f));
}
__device__ __forceinline__ v8f wmma_bf16(v16bf a, v16bf b, v8f c) {
  return __builtin_amdgcn_wmma_f32_16x16x32_bf16(false, a, false, b, (short)0, c, false, false);
}
// A fragment: 16x32 tile from LDS rows (row stride ldk bf16 elements, 16B aligned)
__device__ __forceinline__ v16bf load_a_frag(const unsigned short* lds_base, int ldk, int k0) {
  int lane = threadIdx.x & 31;
  int m = lane & 15, kh = lane >> 4;
  const u32x4* p = (const u32x4*)(lds_base + m * ldk + k0 + kh * 8);
  FragBF f;
  f.u[0] = p[0];   // elements 0..7  -> K = k0 + kh*8 + 0..7
  f.u[1] = p[2];   // elements 8..15 -> K = k0 + 16 + kh*8 + 0..7
  return f.v;
}
// B fragment: pre-swizzled buffer, tile = 512 bf16, lane-contiguous
__device__ __forceinline__ v16bf load_b_frag(const unsigned short* fragbuf, int tile) {
  int lane = threadIdx.x & 31;
  const u32x4* p = (const u32x4*)(fragbuf + tile * 512 + lane * 16);
  FragBF f;
  f.u[0] = p[0];
  f.u[1] = p[1];
  return f.v;
}

// ---------------- ori grid ----------------
__global__ void k_init_ori(float* ori) {
  int i = threadIdx.x;
  if (i < 20) {
    double th = fmod(3.14159265358979323846 * (double)i * (1.0 + sqrt(5.0)),
                     2.0 * 3.14159265358979323846);
    double ph = acos(1.0 - 2.0 * ((double)i + 0.5) / 20.0);
    ori[i * 3 + 0] = (float)(cos(th) * sin(ph));
    ori[i * 3 + 1] = (float)(sin(th) * sin(ph));
    ori[i * 3 + 2] = (float)cos(ph);
  }
}

// ---------------- kNN (top-8 smallest d2, tie-break lower index) ----------------
__global__ void k_knn(const float* x, int* idx) {
  __shared__ unsigned long long key[256];
  int i = blockIdx.x, j = threadIdx.x;
  float dx = x[i * 3 + 0] - x[j * 3 + 0];
  float dy = x[i * 3 + 1] - x[j * 3 + 1];
  float dz = x[i * 3 + 2] - x[j * 3 + 2];
  float d2 = dx * dx + dy * dy + dz * dz;
  key[j] = (((unsigned long long)__float_as_uint(d2)) << 32) | (unsigned int)j;
  __syncthreads();
  if (j == 0) {
    for (int k = 0; k < 8; ++k) {
      unsigned long long best = key[0]; int bi = 0;
      for (int t = 1; t < 256; ++t) if (key[t] < best) { best = key[t]; bi = t; }
      idx[i * 8 + k] = (int)(best & 0xffffffffu);
      key[bi] = ~0ULL;
    }
  }
}

// ---------------- embed: h0 = xn @ We broadcast over O; gather pn ----------------
__global__ void k_embed(const float* x, const float* pos, const int* idx,
                        const float* We, float* h, float* pn) {
  int p = blockIdx.x, c = threadIdx.x;   // 128 threads
  for (int k = 0; k < 8; ++k) {
    int j = idx[p * 8 + k];
    float e = x[j * 3 + 0] * We[0 * 128 + c] + x[j * 3 + 1] * We[1 * 128 + c] +
              x[j * 3 + 2] * We[2 * 128 + c];
    for (int o = 0; o < 20; ++o) h[((p * 8 + k) * 20 + o) * 128 + c] = e;
    if (c < 3) pn[(p * 8 + k) * 3 + c] = pos[j * 3 + c];
  }
}

__global__ void k_zero(float* buf, int n) {
  int i = blockIdx.x * blockDim.x + threadIdx.x;
  if (i < n) buf[i] = 0.0f;
}

// ---------------- fk[l] = gelu(gelu(poly(oi)@Wf1+bf1)@Wf2+bf2) @ Wfk[l] ----------------
__global__ void k_fk(const float* ori, const float* Wf1, const float* bf1,
                     const float* Wf2, const float* bf2, const float* Wfk, float* fk) {
  __shared__ float hid[128], fbv[128];
  int po = blockIdx.x / 20, o = blockIdx.x % 20, c = threadIdx.x;
  float t = ori[po * 3] * ori[o * 3] + ori[po * 3 + 1] * ori[o * 3 + 1] +
            ori[po * 3 + 2] * ori[o * 3 + 2];
  float t2 = t * t, t3 = t2 * t;
  hid[c] = gelu_f(t * Wf1[c] + t2 * Wf1[128 + c] + t3 * Wf1[256 + c] + bf1[c]);
  __syncthreads();
  float a = bf2[c];
  for (int k2 = 0; k2 < 128; ++k2) a += hid[k2] * Wf2[k2 * 128 + c];
  fbv[c] = gelu_f(a);
  __syncthreads();
  for (int l = 0; l < 2; ++l) {
    float s = 0.0f;
    for (int k2 = 0; k2 < 128; ++k2) s += fbv[k2] * Wfk[(l * 128 + k2) * 128 + c];
    fk[((l * 20 + po) * 20 + o) * 128 + c] = s;
  }
}

// ---------------- repack f32 weight (Kact x N) into WMMA-B bf16 fragments ----------------
__global__ void k_pack_b(const float* W, int Kact, int N, unsigned short* dst) {
  int ntiles = N >> 4;
  int t = blockIdx.x;
  int kt = t / ntiles, nt = t % ntiles;
  int lane = threadIdx.x;
  int kb = kt * 32 + (lane >> 4) * 8;
  int n = nt * 16 + (lane & 15);
  unsigned short* d = dst + t * 512 + lane * 16;
  for (int e = 0; e < 16; ++e) {
    int k = kb + (e < 8 ? e : 8 + e);
    d[e] = (k < Kact) ? f2bf(W[k * N + n]) : (unsigned short)0;
  }
}

// ---------------- fused: spatial -> poly -> W1 -> gelu -> W2 -> gelu (kb) -> Wk[l]
//                  -> sum_n kern * h  => x1[p,m,o,c].  Block = (p,m), 2 waves. ----------
__global__ __launch_bounds__(64) void k_x1(
    const float* __restrict__ pn, const float* __restrict__ ori,
    const float* __restrict__ h,
    const unsigned short* __restrict__ W1f, const float* __restrict__ b1,
    const unsigned short* __restrict__ W2f, const float* __restrict__ b2,
    const unsigned short* __restrict__ Wkf, float* __restrict__ x1) {
  __shared__ __align__(16) unsigned short polyA[2][16 * 32];
  __shared__ __align__(16) unsigned short actA[2][16 * 128];
  __shared__ __align__(16) unsigned short kbA[2][16 * 128];
  __shared__ float oris[60], pns[24];
  int p = blockIdx.x >> 3, m = blockIdx.x & 7;
  int wv = threadIdx.x >> 5, lane = threadIdx.x & 31;
  if (threadIdx.x < 60) oris[threadIdx.x] = ori[threadIdx.x];
  if (threadIdx.x < 24) pns[threadIdx.x] = pn[p * 24 + threadIdx.x];
  __syncthreads();
  float pmx = pns[m * 3], pmy = pns[m * 3 + 1], pmz = pns[m * 3 + 2];
  unsigned short* myPoly = polyA[wv];
  unsigned short* myAct = actA[wv];
  unsigned short* myKb = kbA[wv];
  int cl = lane & 15, ch = lane >> 4;

  for (int t = wv; t < 10; t += 2) {   // 10 row-tiles of 16 rows = (o pair) x (n=0..7)
    if (lane < 16) {
      int r = lane;
      int o = 2 * t + (r >> 3);
      int n = r & 7;
      float rx = pns[n * 3] - pmx, ry = pns[n * 3 + 1] - pmy, rz = pns[n * 3 + 2] - pmz;
      float a = rx * oris[o * 3] + ry * oris[o * 3 + 1] + rz * oris[o * 3 + 2];
      float nrm2 = rx * rx + ry * ry + rz * rz;
      float om = 1.0f - a;
      float b = sqrtf(nrm2 * om * om + 1e-12f);
      float f[14];
      f[0] = a; f[1] = b;
      f[2] = a * a; f[3] = a * b; f[4] = b * a; f[5] = b * b;
      f[6] = f[2] * a; f[7] = f[2] * b; f[8] = f[3] * a; f[9] = f[3] * b;
      f[10] = f[4] * a; f[11] = f[4] * b; f[12] = f[5] * a; f[13] = f[5] * b;
      #pragma unroll
      for (int e = 0; e < 14; ++e) myPoly[r * 32 + e] = f2bf(f[e]);
      #pragma unroll
      for (int e = 14; e < 32; ++e) myPoly[r * 32 + e] = 0;
    }
    // matmul1: poly(16x32) @ W1(32x128)
    v8f acc[8] = {};
    v16bf aP = load_a_frag(myPoly, 32, 0);
    #pragma unroll
    for (int nc = 0; nc < 8; ++nc) acc[nc] = wmma_bf16(aP, load_b_frag(W1f, nc), acc[nc]);
    #pragma unroll
    for (int nc = 0; nc < 8; ++nc) {
      int col = nc * 16 + cl;
      float bb = b1[col];
      #pragma unroll
      for (int v = 0; v < 8; ++v)
        myAct[(v + 8 * ch) * 128 + col] = f2bf(gelu_f(acc[nc][v] + bb));
    }
    // matmul2: act(16x128) @ W2(128x128) -> kb
    v8f acc2[8] = {};
    for (int kc = 0; kc < 4; ++kc) {
      v16bf af = load_a_frag(myAct, 128, kc * 32);
      #pragma unroll
      for (int nc = 0; nc < 8; ++nc)
        acc2[nc] = wmma_bf16(af, load_b_frag(W2f, kc * 8 + nc), acc2[nc]);
    }
    #pragma unroll
    for (int nc = 0; nc < 8; ++nc) {
      int col = nc * 16 + cl;
      float bb = b2[col];
      #pragma unroll
      for (int v = 0; v < 8; ++v)
        myKb[(v + 8 * ch) * 128 + col] = f2bf(gelu_f(acc2[nc][v] + bb));
    }
    // matmul3: kb(16x128) @ Wk[l](128x128) -> kern
    v8f acc3[8] = {};
    for (int kc = 0; kc < 4; ++kc) {
      v16bf af = load_a_frag(myKb, 128, kc * 32);
      #pragma unroll
      for (int nc = 0; nc < 8; ++nc)
        acc3[nc] = wmma_bf16(af, load_b_frag(Wkf, kc * 8 + nc), acc3[nc]);
    }
    // reduction over n (the 8 D-fragment rows of this lane's half) with h
    int o = 2 * t + ch;
    #pragma unroll
    for (int nc = 0; nc < 8; ++nc) {
      int col = nc * 16 + cl;
      float s = 0.0f;
      #pragma unroll
      for (int v = 0; v < 8; ++v)
        s += acc3[nc][v] * h[((p * 8 + v) * 20 + o) * 128 + col];
      x1[((p * 8 + m) * 20 + o) * 128 + col] = s;
    }
  }
}

// ---------------- fused: x2 einsum over o -> LayerNorm -> MLP(128->512->128) WMMA
//                  -> residual h += z -> readout h@Wr WMMA. Block = (p,m), 2 waves. ------
__global__ __launch_bounds__(64) void k_x2mlp(
    const float* __restrict__ x1, const float* __restrict__ fk_l,
    const float* __restrict__ cbias_l, const float* __restrict__ gamma_l,
    const float* __restrict__ beta_l,
    const unsigned short* __restrict__ Wb1f_l, const float* __restrict__ bb1_l,
    const unsigned short* __restrict__ Wb2f_l, const float* __restrict__ bb2_l,
    const unsigned short* __restrict__ Wrf_l, const float* __restrict__ br_l,
    float* __restrict__ h, float* __restrict__ readouts) {
  __shared__ __align__(16) float x1s[20 * 128];
  __shared__ __align__(16) float x2s[20 * 128];
  __shared__ __align__(16) unsigned short zA[32 * 128];
  __shared__ __align__(16) unsigned short uA[32 * 512];
  __shared__ float mu[20], iv[20];
  int p = blockIdx.x >> 3, m = blockIdx.x & 7;
  int tid = threadIdx.x, wv = tid >> 5, lane = tid & 31;
  int cl = lane & 15, ch = lane >> 4;
  const float* x1p = x1 + (size_t)((p * 8 + m) * 20) * 128;
  for (int e = tid; e < 2560; e += 64) x1s[e] = x1p[e];
  for (int e = tid; e < 12 * 128; e += 64) zA[20 * 128 + e] = 0;  // pad rows 20..31
  __syncthreads();
  // x2[po,c] = (sum_o x1[o,c]*fk[po,o,c]) / H + cbias[c]
  for (int e = tid; e < 2560; e += 64) {
    int po = e >> 7, c = e & 127;
    float s = 0.0f;
    for (int o = 0; o < 20; ++o) s += x1s[o * 128 + c] * fk_l[(po * 20 + o) * 128 + c];
    x2s[e] = s * (1.0f / 128.0f) + cbias_l[c];
  }
  __syncthreads();
  if (tid < 20) {
    float s = 0.0f;
    for (int c = 0; c < 128; ++c) s += x2s[tid * 128 + c];
    float mean = s * (1.0f / 128.0f);
    float vs = 0.0f;
    for (int c = 0; c < 128; ++c) { float d = x2s[tid * 128 + c] - mean; vs += d * d; }
    mu[tid] = mean;
    iv[tid] = rsqrtf(vs * (1.0f / 128.0f) + 1e-5f);
  }
  __syncthreads();
  for (int e = tid; e < 2560; e += 64) {
    int po = e >> 7, c = e & 127;
    zA[e] = f2bf((x2s[e] - mu[po]) * iv[po] * gamma_l[c] + beta_l[c]);
  }
  __syncthreads();
  // MLP matmul1: z(16x128) @ Wb1(128x512), wave wv owns tile wv (rows wv*16..)
  const unsigned short* zbase = zA + wv * 16 * 128;
  for (int ng = 0; ng < 4; ++ng) {
    v8f acc[8] = {};
    for (int kc = 0; kc < 4; ++kc) {
      v16bf af = load_a_frag(zbase, 128, kc * 32);
      #pragma unroll
      for (int nc = 0; nc < 8; ++nc)
        acc[nc] = wmma_bf16(af, load_b_frag(Wb1f_l, kc * 32 + ng * 8 + nc), acc[nc]);
    }
    #pragma unroll
    for (int nc = 0; nc < 8; ++nc) {
      int col = ng * 128 + nc * 16 + cl;
      float bb = bb1_l[col];
      #pragma unroll
      for (int v = 0; v < 8; ++v)
        uA[(wv * 16 + v + 8 * ch) * 512 + col] = f2bf(gelu_f(acc[nc][v] + bb));
    }
  }
  // MLP matmul2: u(16x512) @ Wb2(512x128)
  v8f acc2[8] = {};
  const unsigned short* ubase = uA + wv * 16 * 512;
  for (int kc = 0; kc < 16; ++kc) {
    v16bf af = load_a_frag(ubase, 512, kc * 32);
    #pragma unroll
    for (int nc = 0; nc < 8; ++nc)
      acc2[nc] = wmma_bf16(af, load_b_frag(Wb2f_l, kc * 8 + nc), acc2[nc]);
  }
  // residual update h, stage new h (bf16) into zA for readout
  float* hp = h + (size_t)((p * 8 + m) * 20) * 128;
  #pragma unroll
  for (int nc = 0; nc < 8; ++nc) {
    int col = nc * 16 + cl;
    float bb = bb2_l[col];
    #pragma unroll
    for (int v = 0; v < 8; ++v) {
      int row = wv * 16 + v + 8 * ch;
      if (row < 20) {
        float hn = hp[row * 128 + col] + acc2[nc][v] + bb;
        hp[row * 128 + col] = hn;
        zA[row * 128 + col] = f2bf(hn);
      }
    }
  }
  // readout: h(16x128) @ Wr(128x32), accumulate into readouts
  v8f acc3[2] = {};
  for (int kc = 0; kc < 4; ++kc) {
    v16bf af = load_a_frag(zbase, 128, kc * 32);
    #pragma unroll
    for (int nc = 0; nc < 2; ++nc)
      acc3[nc] = wmma_bf16(af, load_b_frag(Wrf_l, kc * 2 + nc), acc3[nc]);
  }
  float* rp = readouts + (size_t)((p * 8 + m) * 20) * 32;
  #pragma unroll
  for (int nc = 0; nc < 2; ++nc) {
    int col = nc * 16 + cl;
    float bb = br_l[col];
    #pragma unroll
    for (int v = 0; v < 8; ++v) {
      int row = wv * 16 + v + 8 * ch;
      if (row < 20) rp[row * 32 + col] += acc3[nc][v] + bb;
    }
  }
}

// ---------------- final means ----------------
__global__ void k_final(const float* readouts, const float* ori, float* out) {
  int p = blockIdx.x, t = threadIdx.x;  // 64 threads
  const float* rp = readouts + (size_t)p * 8 * 20 * 32;
  if (t < 16) {
    float s = 0.0f;
    for (int m = 0; m < 8; ++m)
      for (int o = 0; o < 20; ++o) s += rp[(m * 20 + o) * 32 + t];
    out[p * 16 + t] = s * (1.0f / 320.0f);   // /L /O /K = /(2*20*8)
  } else if (t < 64) {
    int q = t - 16; int c = q / 3, d = q % 3;
    float s = 0.0f;
    for (int m = 0; m < 8; ++m)
      for (int o = 0; o < 20; ++o)
        s += rp[(m * 20 + o) * 32 + 16 + c] * ori[o * 3 + d];
    out[4096 + (p * 16 + c) * 3 + d] = s * (1.0f / 320.0f);
  }
}

extern "C" void kernel_launch(void* const* d_in, const int* in_sizes, int n_in,
                              void* d_out, int out_size, void* d_ws, size_t ws_size,
                              hipStream_t stream) {
  (void)in_sizes; (void)n_in; (void)out_size; (void)ws_size;
  const float* x     = (const float*)d_in[0];
  const float* pos   = (const float*)d_in[1];
  const float* W1    = (const float*)d_in[2];
  const float* b1    = (const float*)d_in[3];
  const float* W2    = (const float*)d_in[4];
  const float* b2    = (const float*)d_in[5];
  const float* Wf1   = (const float*)d_in[6];
  const float* bf1   = (const float*)d_in[7];
  const float* Wf2   = (const float*)d_in[8];
  const float* bf2   = (const float*)d_in[9];
  const float* We    = (const float*)d_in[10];
  const float* Wk    = (const float*)d_in[11];
  const float* Wfk   = (const float*)d_in[12];
  const float* cbias = (const float*)d_in[13];
  const float* gamma = (const float*)d_in[14];
  const float* beta  = (const float*)d_in[15];
  const float* Wb1   = (const float*)d_in[16];
  const float* bb1   = (const float*)d_in[17];
  const float* Wb2   = (const float*)d_in[18];
  const float* bb2   = (const float*)d_in[19];
  const float* Wr    = (const float*)d_in[20];
  const float* br    = (const float*)d_in[21];

  char* ws = (char*)d_ws;
  size_t off = 0;
  auto alloc = [&](size_t bytes) -> char* {
    char* q = ws + off;
    off = (off + bytes + 255) & ~(size_t)255;
    return q;
  };
  float* ori  = (float*)alloc(20 * 3 * 4);
  int*   idx  = (int*)alloc(256 * 8 * 4);
  float* pn   = (float*)alloc(256 * 8 * 3 * 4);
  float* h    = (float*)alloc((size_t)256 * 8 * 20 * 128 * 4);
  float* x1b  = (float*)alloc((size_t)256 * 8 * 20 * 128 * 4);
  float* rout = (float*)alloc((size_t)256 * 8 * 20 * 32 * 4);
  float* fk   = (float*)alloc((size_t)2 * 20 * 20 * 128 * 4);
  unsigned short* W1f  = (unsigned short*)alloc(8 * 512 * 2);
  unsigned short* W2f  = (unsigned short*)alloc(32 * 512 * 2);
  unsigned short* Wkf  = (unsigned short*)alloc(2 * 32 * 512 * 2);
  unsigned short* Wb1f = (unsigned short*)alloc((size_t)2 * 128 * 512 * 2);
  unsigned short* Wb2f = (unsigned short*)alloc((size_t)2 * 128 * 512 * 2);
  unsigned short* Wrf  = (unsigned short*)alloc(2 * 8 * 512 * 2);

  k_init_ori<<<1, 32, 0, stream>>>(ori);
  k_knn<<<256, 256, 0, stream>>>(x, idx);
  k_embed<<<256, 128, 0, stream>>>(x, pos, idx, We, h, pn);
  int nr = 256 * 8 * 20 * 32;
  k_zero<<<(nr + 255) / 256, 256, 0, stream>>>(rout, nr);
  k_fk<<<400, 128, 0, stream>>>(ori, Wf1, bf1, Wf2, bf2, Wfk, fk);
  k_pack_b<<<8, 32, 0, stream>>>(W1, 14, 128, W1f);
  k_pack_b<<<32, 32, 0, stream>>>(W2, 128, 128, W2f);
  for (int l = 0; l < 2; ++l) {
    k_pack_b<<<32, 32, 0, stream>>>(Wk + (size_t)l * 128 * 128, 128, 128, Wkf + (size_t)l * 32 * 512);
    k_pack_b<<<128, 32, 0, stream>>>(Wb1 + (size_t)l * 128 * 512, 128, 512, Wb1f + (size_t)l * 128 * 512);
    k_pack_b<<<128, 32, 0, stream>>>(Wb2 + (size_t)l * 512 * 128, 512, 128, Wb2f + (size_t)l * 128 * 512);
    k_pack_b<<<8, 32, 0, stream>>>(Wr + (size_t)l * 128 * 32, 128, 32, Wrf + (size_t)l * 8 * 512);
  }
  for (int l = 0; l < 2; ++l) {
    k_x1<<<2048, 64, 0, stream>>>(pn, ori, h, W1f, b1, W2f, b2,
                                  Wkf + (size_t)l * 32 * 512, x1b);
    k_x2mlp<<<2048, 64, 0, stream>>>(
        x1b, fk + (size_t)l * 20 * 20 * 128, cbias + l * 128, gamma + l * 128,
        beta + l * 128, Wb1f + (size_t)l * 128 * 512, bb1 + l * 512,
        Wb2f + (size_t)l * 128 * 512, bb2 + l * 128, Wrf + (size_t)l * 8 * 512,
        br + l * 32, h, rout);
  }
  k_final<<<256, 64, 0, stream>>>(rout, ori, (float*)d_out);
}
